// Transformer_40235253628956
// MI455X (gfx1250) — matrix operand order repaired and checked by reading, not compiled, not run on hardware
//
#include <hip/hip_runtime.h>
#include <hip/hip_bf16.h>
#include <math.h>
#include <stdint.h>

// ---------------- model constants ----------------
#define D_    1024
#define F_    4096
#define H_    8
#define HD_   128
#define NE_   4
#define NDL_  4
#define B_    4
#define L_    1024
#define NTOK_ (B_*L_)
#define EPS_  1e-5f
#define SCALE_ 0.03125f            // 1/sqrt(1024)

typedef __bf16 bf16;
typedef __attribute__((ext_vector_type(16))) __bf16 v16bf;
typedef __attribute__((ext_vector_type(8)))  __bf16 v8bf;
typedef __attribute__((ext_vector_type(8)))  float  v8f;
typedef __attribute__((vector_size(16)))     int    v4i_;

// ---------------- gfx1250 async global->LDS copy (guarded) ----------------
#if defined(__gfx1250__) && __has_builtin(__builtin_amdgcn_global_load_async_to_lds_b128) && __has_builtin(__builtin_amdgcn_s_wait_asynccnt)
#define ASYNC_LDS 1
#else
#define ASYNC_LDS 0
#endif

__device__ __forceinline__ void async_copy16(const bf16* g, bf16* l)
{
#if ASYNC_LDS
    __builtin_amdgcn_global_load_async_to_lds_b128(
        (__attribute__((address_space(1))) v4i_*)(uintptr_t)g,
        (__attribute__((address_space(3))) v4i_*)(uint32_t)(uintptr_t)l, 0, 0);
#else
    *(v8bf*)l = *(const v8bf*)g;
#endif
}
__device__ __forceinline__ void wait_async()
{
#if ASYNC_LDS
    __builtin_amdgcn_s_wait_asynccnt(0);
#endif
}

// =====================================================================
// bf16 WMMA GEMM:  C[z] = alpha * A[z] * B[z] (+bias)(+res)(relu)
// A: bf16 [M][K] (lda).  B: bf16 [N][K] if BT (contiguous along K;
// weights are pre-transposed, K-matrix naturally so), else [K][N] (V).
// Block tile 128(M) x 128(N), BK=32.  8 waves: wm=w>>1 (32-row strip,
// 2 A-frags), wn=w&1 (64-col strip, 4 B-frags) -> 8 WMMA / wave / K-step.
// blockIdx.z = b*H+h for attention slices via strides.
// =====================================================================
template<bool BT>
__global__ __launch_bounds__(256)
void gemm_bf16_wmma(const bf16* __restrict__ A, int lda, long long aBS, long long aHS,
                    const bf16* __restrict__ Bm, int ldb, long long bBS, long long bHS,
                    float* __restrict__ Cf, bf16* __restrict__ Cb,
                    int ldc, long long cBS, long long cHS,
                    const float* __restrict__ bias, const float* __restrict__ res,
                    int K, float alpha, int relu)
{
    constexpr int LDK = 40;                  // padded K-stride (bf16): 80B rows, 16B aligned
    __shared__ bf16 Als[128 * LDK];
    __shared__ bf16 Bls[128 * LDK];

    const int tid  = threadIdx.x;
    const int lane = tid & 31;
    const int w    = tid >> 5;
    const int wm   = w >> 1;
    const int wn   = w & 1;
    const int l15  = lane & 15;
    const int hi   = lane >> 4;

    const int bz = blockIdx.z;
    const int bb = bz / H_;
    const int hh = bz % H_;
    const bf16* Ap = A  + (long long)bb * aBS + (long long)hh * aHS;
    const bf16* Bp = Bm + (long long)bb * bBS + (long long)hh * bHS;
    float* Cfp = Cf ? Cf + (long long)bb * cBS + (long long)hh * cHS : nullptr;
    bf16*  Cbp = Cb ? Cb + (long long)bb * cBS + (long long)hh * cHS : nullptr;

    const int row0 = blockIdx.y * 128;
    const int col0 = blockIdx.x * 128;

    v8f acc[2][4];
#pragma unroll
    for (int u = 0; u < 2; ++u)
#pragma unroll
        for (int t = 0; t < 4; ++t) acc[u][t] = (v8f)0.f;

    const int sr = tid >> 1;                 // staging row 0..127
    const int sc = (tid & 1) << 4;           // staging col 0 or 16 (bf16 units)

    for (int k0 = 0; k0 < K; k0 += 32) {
        // ---- stage A tile (128x32 bf16): one async b128 per thread
        async_copy16(Ap + (long long)(row0 + sr) * lda + k0 + sc, &Als[sr * LDK + sc]);
        if (BT) {
            async_copy16(Bp + (long long)(col0 + sr) * ldb + k0 + sc, &Bls[sr * LDK + sc]);
        } else {
            // B stored [K][N]: read 8 bf16 along n, transpose into Bls[n][k]
#pragma unroll
            for (int it = 0; it < 2; ++it) {
                const int idx = tid + it * 256;      // 32 k * 16 chunks
                const int kk  = idx >> 4;
                const int n8  = (idx & 15) << 3;
                const v8bf vv = *(const v8bf*)(Bp + (long long)(k0 + kk) * ldb + col0 + n8);
#pragma unroll
                for (int j = 0; j < 8; ++j) Bls[(n8 + j) * LDK + kk] = vv[j];
            }
        }
        if (k0 + 32 < K) {
            __builtin_prefetch(Ap + (long long)(row0 + sr) * lda + k0 + 32 + sc, 0, 1);
            if (BT) __builtin_prefetch(Bp + (long long)(col0 + sr) * ldb + k0 + 32 + sc, 0, 1);
        }
        wait_async();
        __syncthreads();

        // ---- A fragments: lanes0-15 K={0..7,16..23}, lanes16-31 K={8..15,24..31}
        union frag_t { v16bf v; v8bf h[2]; };
        frag_t a[2];
        const int ka = hi ? 8 : 0;
#pragma unroll
        for (int u = 0; u < 2; ++u) {
            const int m = wm * 32 + u * 16 + l15;
            a[u].h[0] = *(const v8bf*)&Als[m * LDK + ka];
            a[u].h[1] = *(const v8bf*)&Als[m * LDK + ka + 16];
        }
        // ---- B fragments: lanes0-15 K={0..7,16..23}, lanes16-31 K={8..15,24..31}
        const int kb = hi ? 8 : 0;
#pragma unroll
        for (int t = 0; t < 4; ++t) {
            frag_t bfr;
            const int n = wn * 64 + t * 16 + l15;
            bfr.h[0] = *(const v8bf*)&Bls[n * LDK + kb];
            bfr.h[1] = *(const v8bf*)&Bls[n * LDK + kb + 16];
#pragma unroll
            for (int u = 0; u < 2; ++u)
                acc[u][t] = __builtin_amdgcn_wmma_f32_16x16x32_bf16(
                    false, a[u].v, false, bfr.v, (short)0, acc[u][t], false, false);
        }
        __syncthreads();
    }

    // ---- epilogue: D VGPR r -> (m = r + 8*hi, n = lane&15)
#pragma unroll
    for (int u = 0; u < 2; ++u)
#pragma unroll
    for (int t = 0; t < 4; ++t) {
        const int n = col0 + wn * 64 + t * 16 + l15;
#pragma unroll
        for (int r = 0; r < 8; ++r) {
            const int m = row0 + wm * 32 + u * 16 + r + hi * 8;
            float val = acc[u][t][r] * alpha;
            if (bias) val += bias[n];
            if (res)  val += res[(long long)m * ldc + n];
            if (relu) val = fmaxf(val, 0.f);
            const long long off = (long long)m * ldc + n;
            if (Cfp) Cfp[off] = val;
            if (Cbp) Cbp[off] = (bf16)val;
        }
    }
}

// =====================================================================
// Weight convert + transpose:  W fp32 [K][N]  ->  Wt bf16 [N][K]
// =====================================================================
__global__ __launch_bounds__(256)
void wcvt_t_kernel(const float* __restrict__ W, bf16* __restrict__ Wt, int K, int N)
{
    __shared__ bf16 T[32][33];
    const int n0 = blockIdx.x * 32, k0 = blockIdx.y * 32;
    const int tx = threadIdx.x & 31, ty = threadIdx.x >> 5;
#pragma unroll
    for (int i = 0; i < 4; ++i) {
        const int r = ty + i * 8;
        T[tx][r] = (bf16)W[(long long)(k0 + r) * N + n0 + tx];
    }
    __syncthreads();
#pragma unroll
    for (int i = 0; i < 4; ++i) {
        const int r = ty + i * 8;
        Wt[(long long)(n0 + r) * K + k0 + tx] = T[r][tx];
    }
}

// =====================================================================
// LayerNorm (sqrt(var_unbiased)+eps, as in reference); fp32 out + bf16 copy
// =====================================================================
__global__ __launch_bounds__(256)
void layernorm_kernel(const float* __restrict__ X, const float* __restrict__ g,
                      const float* __restrict__ b, float* __restrict__ Yf,
                      bf16* __restrict__ Yb)
{
    __shared__ float red[18];
    const long long row = blockIdx.x;
    const float* x = X + row * D_;
    float s = 0.f, ss = 0.f;
    for (int i = threadIdx.x; i < D_; i += 256) { const float v = x[i]; s += v; ss += v * v; }
    for (int off = 16; off; off >>= 1) {
        s  += __shfl_down(s,  off, 32);
        ss += __shfl_down(ss, off, 32);
    }
    const int lane = threadIdx.x & 31, wv = threadIdx.x >> 5;
    if (lane == 0) { red[wv] = s; red[wv + 8] = ss; }
    __syncthreads();
    if (threadIdx.x == 0) {
        float a = 0.f, c = 0.f;
        for (int k = 0; k < 8; ++k) { a += red[k]; c += red[k + 8]; }
        red[16] = a; red[17] = c;
    }
    __syncthreads();
    const float mean = red[16] / D_;
    const float var  = (red[17] - D_ * mean * mean) / (float)(D_ - 1);
    const float inv  = 1.f / (sqrtf(var) + EPS_);
    for (int i = threadIdx.x; i < D_; i += 256) {
        const float o = g[i] * (x[i] - mean) * inv + b[i];
        Yf[row * D_ + i] = o;
        if (Yb) Yb[row * D_ + i] = (bf16)o;
    }
}

// =====================================================================
// Softmax: reads fp32 scores, writes bf16 probs (register resident, 4/thr)
// =====================================================================
__global__ __launch_bounds__(256)
void softmax_kernel(const float* __restrict__ S, bf16* __restrict__ P, int causal)
{
    __shared__ float red[10];
    const long long row = blockIdx.x;
    const int i   = (int)(row & (L_ - 1));
    const int lim = causal ? (i + 1) : L_;
    const float* s = S + row * (long long)L_;
    bf16* p = P + row * (long long)L_;
    const int lane = threadIdx.x & 31, wv = threadIdx.x >> 5;

    float v[4];
#pragma unroll
    for (int it = 0; it < 4; ++it) {
        const int j = threadIdx.x + it * 256;
        v[it] = (j < lim) ? s[j] : -INFINITY;
    }
    float mx = fmaxf(fmaxf(v[0], v[1]), fmaxf(v[2], v[3]));
    for (int off = 16; off; off >>= 1) mx = fmaxf(mx, __shfl_down(mx, off, 32));
    if (lane == 0) red[wv] = mx;
    __syncthreads();
    if (threadIdx.x == 0) {
        float m = red[0];
        for (int k = 1; k < 8; ++k) m = fmaxf(m, red[k]);
        red[8] = m;
    }
    __syncthreads();
    mx = red[8];

    float e[4], sum = 0.f;
#pragma unroll
    for (int it = 0; it < 4; ++it) { e[it] = __expf(v[it] - mx); sum += e[it]; }
    for (int off = 16; off; off >>= 1) sum += __shfl_down(sum, off, 32);
    __syncthreads();
    if (lane == 0) red[wv] = sum;
    __syncthreads();
    if (threadIdx.x == 0) {
        float a = 0.f;
        for (int k = 0; k < 8; ++k) a += red[k];
        red[9] = a;
    }
    __syncthreads();
    const float inv = 1.f / red[9];
#pragma unroll
    for (int it = 0; it < 4; ++it)
        p[threadIdx.x + it * 256] = (bf16)(e[it] * inv);
}

// =====================================================================
// Embedding * sqrt(D) + sinusoidal PE; fp32 + bf16 outputs
// =====================================================================
__global__ __launch_bounds__(256)
void embed_pe_kernel(const int* __restrict__ tok, const float* __restrict__ emb,
                     float* __restrict__ outf, bf16* __restrict__ outb)
{
    const long long idx = (long long)blockIdx.x * 256 + threadIdx.x;
    const int n = (int)(idx >> 10);
    const int d = (int)(idx & (D_ - 1));
    const int i = n & (L_ - 1);
    const int t = tok[n];
    float val = emb[(long long)t * D_ + d] * 32.0f;          // sqrt(1024)
    const int k2 = d & ~1;
    const float dv = __expf((float)k2 * (-9.210340371976184f / (float)D_));
    const float ang = (float)i * dv;
    val += (d & 1) ? __cosf(ang) : __sinf(ang);
    outf[idx] = val;
    outb[idx] = (bf16)val;
}

// =====================================================================
// Host-side orchestration
// =====================================================================
static void gemm_t(hipStream_t st, const bf16* A, int lda, const bf16* Bt, int ldb,
                   float* Cf, bf16* Cb, int ldc, const float* bias, const float* res,
                   int M, int N, int K, int relu)
{
    dim3 g(N / 128, M / 128, 1), b(256, 1, 1);
    gemm_bf16_wmma<true><<<g, b, 0, st>>>(A, lda, 0, 0, Bt, ldb, 0, 0,
                                          Cf, Cb, ldc, 0, 0, bias, res, K, 1.f, relu);
}

static bf16* cvtw(hipStream_t st, const float* W, int K, int N, bf16*& cur)
{
    bf16* dst = cur; cur += (size_t)K * N;
    dim3 g(N / 32, K / 32, 1), b(256, 1, 1);
    wcvt_t_kernel<<<g, b, 0, st>>>(W, dst, K, N);
    return dst;
}

static void run_mha(hipStream_t st, const bf16* kvb, const bf16* qinb,
                    const bf16* Wq, const bf16* Wk, const bf16* Wv, const bf16* Wo,
                    const float* bq, const float* bk, const float* bv, const float* bo,
                    const float* resid, bf16* qb, bf16* kb, bf16* vb, bf16* attb,
                    float* sc, bf16* pb, float* outf, int causal)
{
    gemm_t(st, qinb, D_, Wq, D_, nullptr, qb, D_, bq, nullptr, NTOK_, D_, D_, 0);
    gemm_t(st, kvb,  D_, Wk, D_, nullptr, kb, D_, bk, nullptr, NTOK_, D_, D_, 0);
    gemm_t(st, kvb,  D_, Wv, D_, nullptr, vb, D_, bv, nullptr, NTOK_, D_, D_, 0);
    {   // scores[z] = scale * Q K^T  (K rows are [N][K]-contiguous along d)
        dim3 g(L_ / 128, L_ / 128, B_ * H_), b(256, 1, 1);
        gemm_bf16_wmma<true><<<g, b, 0, st>>>(
            qb, D_, (long long)L_ * D_, HD_,
            kb, D_, (long long)L_ * D_, HD_,
            sc, nullptr, L_, (long long)H_ * L_ * L_, (long long)L_ * L_,
            nullptr, nullptr, HD_, SCALE_, 0);
    }
    softmax_kernel<<<B_ * H_ * L_, 256, 0, st>>>(sc, pb, causal);
    {   // att[z] = P V   (V is [K][N] -> BT=false transpose staging)
        dim3 g(HD_ / 128, L_ / 128, B_ * H_), b(256, 1, 1);
        gemm_bf16_wmma<false><<<g, b, 0, st>>>(
            pb, L_, (long long)H_ * L_ * L_, (long long)L_ * L_,
            vb, D_, (long long)L_ * D_, (long long)HD_,
            nullptr, attb, D_, (long long)L_ * D_, (long long)HD_,
            nullptr, nullptr, L_, 1.f, 0);
    }
    gemm_t(st, attb, D_, Wo, D_, outf, nullptr, D_, bo, resid, NTOK_, D_, D_, 0);
}

extern "C" void kernel_launch(void* const* d_in, const int* in_sizes, int n_in,
                              void* d_out, int out_size, void* d_ws, size_t ws_size,
                              hipStream_t stream)
{
    (void)in_sizes; (void)n_in; (void)out_size; (void)ws_size;
    const int*   src     = (const int*)d_in[0];
    const int*   target  = (const int*)d_in[1];
    const float* src_emb = (const float*)d_in[2];
    const float* tgt_emb = (const float*)d_in[3];

    const long long NDsz = (long long)NTOK_ * D_;
    const long long NFsz = (long long)NTOK_ * F_;
    const long long SCsz = (long long)B_ * H_ * L_ * L_;

    char* base = (char*)d_ws;
    float* x    = (float*)base; base += NDsz * 4;
    float* y    = (float*)base; base += NDsz * 4;
    float* tmp  = (float*)base; base += NDsz * 4;
    float* sc   = (float*)base; base += SCsz * 4;
    bf16* xb    = (bf16*)base;  base += NDsz * 2;
    bf16* yb    = (bf16*)base;  base += NDsz * 2;
    bf16* qb    = (bf16*)base;  base += NDsz * 2;
    bf16* kbb   = (bf16*)base;  base += NDsz * 2;
    bf16* vbb   = (bf16*)base;  base += NDsz * 2;
    bf16* attb  = (bf16*)base;  base += NDsz * 2;
    bf16* ffnhb = (bf16*)base;  base += NFsz * 2;
    bf16* pb    = (bf16*)base;  base += SCsz * 2;
    bf16* wcur  = (bf16*)base;

    // ---- one-time (per launch) weight convert+transpose to bf16 [N][K]
    const int dec_base = 4 + NE_ * 16;
    bf16* encW[NE_][6];
    for (int e = 0; e < NE_; ++e) {
        const float* const* pp = (const float* const*)(d_in + 4 + e * 16);
        encW[e][0] = cvtw(stream, pp[0],  D_, D_, wcur);   // Wq
        encW[e][1] = cvtw(stream, pp[2],  D_, D_, wcur);   // Wk
        encW[e][2] = cvtw(stream, pp[4],  D_, D_, wcur);   // Wv
        encW[e][3] = cvtw(stream, pp[6],  D_, D_, wcur);   // Wo
        encW[e][4] = cvtw(stream, pp[8],  D_, F_, wcur);   // W1 -> [F][D]
        encW[e][5] = cvtw(stream, pp[10], F_, D_, wcur);   // W2 -> [D][F]
    }
    bf16* decW[NDL_][10];
    for (int e = 0; e < NDL_; ++e) {
        const float* const* pp = (const float* const*)(d_in + dec_base + e * 26);
        decW[e][0] = cvtw(stream, pp[0],  D_, D_, wcur);   // sa Wq
        decW[e][1] = cvtw(stream, pp[2],  D_, D_, wcur);
        decW[e][2] = cvtw(stream, pp[4],  D_, D_, wcur);
        decW[e][3] = cvtw(stream, pp[6],  D_, D_, wcur);
        decW[e][4] = cvtw(stream, pp[8],  D_, D_, wcur);   // ca Wq
        decW[e][5] = cvtw(stream, pp[10], D_, D_, wcur);
        decW[e][6] = cvtw(stream, pp[12], D_, D_, wcur);
        decW[e][7] = cvtw(stream, pp[14], D_, D_, wcur);
        decW[e][8] = cvtw(stream, pp[16], D_, F_, wcur);   // W1
        decW[e][9] = cvtw(stream, pp[18], F_, D_, wcur);   // W2
    }

    const int nblk_emb = (int)(NDsz / 256);

    // ---------------- encoder ----------------
    embed_pe_kernel<<<nblk_emb, 256, 0, stream>>>(src, src_emb, x, xb);
    for (int e = 0; e < NE_; ++e) {
        const float* const* pp = (const float* const*)(d_in + 4 + e * 16);
        run_mha(stream, xb, xb, encW[e][0], encW[e][1], encW[e][2], encW[e][3],
                pp[1], pp[3], pp[5], pp[7], x, qb, kbb, vbb, attb, sc, pb, tmp, 0);
        layernorm_kernel<<<NTOK_, 256, 0, stream>>>(tmp, pp[12], pp[13], x, xb);
        gemm_t(stream, xb, D_, encW[e][4], D_, nullptr, ffnhb, F_, pp[9], nullptr,
               NTOK_, F_, D_, 1);
        gemm_t(stream, ffnhb, F_, encW[e][5], F_, tmp, nullptr, D_, pp[11], x,
               NTOK_, D_, F_, 0);
        layernorm_kernel<<<NTOK_, 256, 0, stream>>>(tmp, pp[14], pp[15], x, xb);
    }

    // ---------------- decoder ----------------
    embed_pe_kernel<<<nblk_emb, 256, 0, stream>>>(target, tgt_emb, y, yb);
    for (int e = 0; e < NDL_; ++e) {
        const float* const* pp = (const float* const*)(d_in + dec_base + e * 26);
        // y = LN1(y + causal-SA(y))
        run_mha(stream, yb, yb, decW[e][0], decW[e][1], decW[e][2], decW[e][3],
                pp[1], pp[3], pp[5], pp[7], y, qb, kbb, vbb, attb, sc, pb, tmp, 1);
        layernorm_kernel<<<NTOK_, 256, 0, stream>>>(tmp, pp[20], pp[21], y, yb);
        // y = LN2(y + CA(enc_out, y))
        run_mha(stream, xb, yb, decW[e][4], decW[e][5], decW[e][6], decW[e][7],
                pp[9], pp[11], pp[13], pp[15], y, qb, kbb, vbb, attb, sc, pb, tmp, 0);
        layernorm_kernel<<<NTOK_, 256, 0, stream>>>(tmp, pp[22], pp[23], y, yb);
        // y = LN3(y + FFN(y))
        gemm_t(stream, yb, D_, decW[e][8], D_, nullptr, ffnhb, F_, pp[17], nullptr,
               NTOK_, F_, D_, 1);
        gemm_t(stream, ffnhb, F_, decW[e][9], F_, tmp, nullptr, D_, pp[19], y,
               NTOK_, D_, F_, 0);
        float* dst = (e == NDL_ - 1) ? (float*)d_out : y;
        bf16*  dstb = (e == NDL_ - 1) ? nullptr : yb;
        layernorm_kernel<<<NTOK_, 256, 0, stream>>>(tmp, pp[24], pp[25], dst, dstb);
    }
}
